// NewCrossEntropyLoss2_43516608643242
// MI455X (gfx1250) — matrix operand order repaired and checked by reading, not compiled, not run on hardware
//
#include <hip/hip_runtime.h>
#include <hip/hip_bf16.h>

#define B_ROWS   512
#define C_COLS   50000
#define D_DIM    256
#define N_IDS    40000
#define CT_TILES (C_COLS / 16)   // 3125
#define LDS_PITCH 264            // 256 + 8 bf16 pad -> 528B row stride, conflict-free

typedef __attribute__((ext_vector_type(16))) __bf16        v16bf;
typedef __attribute__((ext_vector_type(8)))  float         v8f;
typedef __attribute__((ext_vector_type(4)))  unsigned int  v4u;

union FragB { v16bf v; v4u u[2]; };

__device__ __forceinline__ unsigned short f2bf_rne(float x) {
    unsigned u = __float_as_uint(x);
    unsigned r = u + 0x7fffu + ((u >> 16) & 1u);   // round-to-nearest-even
    return (unsigned short)(r >> 16);
}
__device__ __forceinline__ float bf2f(unsigned short h) {
    return __uint_as_float(((unsigned)h) << 16);
}

// ---------------------------------------------------------------- init ------
__global__ __launch_bounds__(256) void init_ws(unsigned char* __restrict__ mask,
                                               float* __restrict__ accums) {
    int i = blockIdx.x * 256 + threadIdx.x;
    if (i < C_COLS)      mask[i]   = 0;
    if (i < 3 * B_ROWS)  accums[i] = 0.0f;
}

__global__ __launch_bounds__(256) void scatter_ids(const int* __restrict__ ids,
                                                   unsigned char* __restrict__ mask) {
    int i = blockIdx.x * 256 + threadIdx.x;
    if (i < N_IDS) mask[ids[i]] = 1;
}

// ------------------------------------------------- visual normalize -> bf16 -
__global__ __launch_bounds__(256) void prep_visual(const float* __restrict__ v,
                                                   unsigned short* __restrict__ v_hi,
                                                   unsigned short* __restrict__ v_lo) {
    int tid = threadIdx.x;
    int row = blockIdx.x * 16 + (tid >> 4);
    int seg = tid & 15;
    const float* src = v + (size_t)row * D_DIM + seg * 16;
    float x[16]; float ss = 0.0f;
#pragma unroll
    for (int i = 0; i < 16; ++i) { x[i] = src[i]; ss += x[i] * x[i]; }
#pragma unroll
    for (int m = 1; m < 16; m <<= 1) ss += __shfl_xor(ss, m, 32);   // within 16-lane group
    float inv = 1.0f / sqrtf(ss);                                    // no eps (visual)
    unsigned short* dh = v_hi + (size_t)row * D_DIM + seg * 16;
    unsigned short* dl = v_lo + (size_t)row * D_DIM + seg * 16;
#pragma unroll
    for (int i = 0; i < 16; ++i) {
        float y = x[i] * inv;
        unsigned short h = f2bf_rne(y);
        dh[i] = h;
        dl[i] = f2bf_rne(y - bf2f(h));
    }
}

// --------------------------------------------------- epilogue helper --------
__device__ __forceinline__ void epilogue_tile(const v8f& acc, int rt, const int* labs,
                                              int half, int n, int ld,
                                              float* __restrict__ denom,
                                              float* __restrict__ sumpos,
                                              float* __restrict__ nposv) {
#pragma unroll
    for (int r = 0; r < 8; ++r) {
        const int mg  = rt * 16 + half * 8 + r;   // C/D layout: M = VGPR index (+8 upper half)
        const int lab = labs[r];
        const float s = acc[r];
        const bool pos = ld && (lab != 0);
        const bool neg = ld && (lab == 0);
        float dn = neg ? expf(s) : 0.0f;
        float ps = pos ? s       : 0.0f;
        float np = pos ? 1.0f    : 0.0f;
#pragma unroll
        for (int m = 8; m >= 1; m >>= 1) {        // reduce across the 16 lanes of this half
            dn += __shfl_xor(dn, m, 32);
            ps += __shfl_xor(ps, m, 32);
            np += __shfl_xor(np, m, 32);
        }
        if (n == 0) {
            atomicAdd(&denom[mg],  dn);
            atomicAdd(&sumpos[mg], ps);
            atomicAdd(&nposv[mg],  np);
        }
    }
}

// ------------------------------------------------------------ main kernel ---
// One block per 16-column text tile; 8 waves; each wave handles 2 row tiles
// per iteration (6 independent WMMA chains) against the register-resident B.
__global__ __launch_bounds__(256) void gemm_loss(
        const unsigned short* __restrict__ v_hi,
        const unsigned short* __restrict__ v_lo,
        const float*          __restrict__ text,
        const int*            __restrict__ labels,
        const unsigned char*  __restrict__ mask,
        float* __restrict__ denom, float* __restrict__ sumpos, float* __restrict__ nposv) {
    __shared__ unsigned short t_hi[16][LDS_PITCH];
    __shared__ unsigned short t_lo[16][LDS_PITCH];

    const int tid = threadIdx.x;
    const int c0  = blockIdx.x * 16;

    // ---- stage: normalize 16 text rows, split bf16 hi/lo into LDS ----
    {
        int j = tid >> 4, seg = tid & 15;
        const float* src = text + (size_t)(c0 + j) * D_DIM + seg * 16;
        float x[16]; float ss = 0.0f;
#pragma unroll
        for (int i = 0; i < 16; ++i) { x[i] = src[i]; ss += x[i] * x[i]; }
#pragma unroll
        for (int m = 1; m < 16; m <<= 1) ss += __shfl_xor(ss, m, 32);
        float inv = 1.0f / (1e-6f + sqrtf(ss));                      // eps on text norm
#pragma unroll
        for (int i = 0; i < 16; ++i) {
            float y = x[i] * inv;
            unsigned short h = f2bf_rne(y);
            t_hi[j][seg * 16 + i] = h;
            t_lo[j][seg * 16 + i] = f2bf_rne(y - bf2f(h));
        }
    }
    __syncthreads();

    const int lane   = tid & 31;
    const int wave   = tid >> 5;
    const int n      = lane & 15;       // C/D: N = lane (mod 16); A: M = lane & 15
    const int half   = lane >> 4;
    const int kbaseA = half * 8;        // A 16x32 bf16 lane layout
    const int kbaseB = half * 8;        // B 32x16 bf16 lane layout (same k order as A)
    const int cg     = c0 + n;
    const int ld     = mask[cg];

    for (int it = 0; it < 2; ++it) {
        const int rt0 = wave + it * 16;     // row tiles rt0 and rt0+8
        const int rt1 = rt0 + 8;
        const size_t arow0 = (size_t)(rt0 * 16 + n) * D_DIM;
        const size_t arow1 = (size_t)(rt1 * 16 + n) * D_DIM;

        // prefetch the label tile rows for both output tiles (overlaps GEMM)
        int labs0[8], labs1[8];
#pragma unroll
        for (int r = 0; r < 8; ++r) {
            labs0[r] = labels[(size_t)(rt0 * 16 + half * 8 + r) * C_COLS + cg];
            labs1[r] = labels[(size_t)(rt1 * 16 + half * 8 + r) * C_COLS + cg];
        }

        v8f x0 = {0,0,0,0,0,0,0,0}, x1 = {0,0,0,0,0,0,0,0}, x2 = {0,0,0,0,0,0,0,0};
        v8f y0 = {0,0,0,0,0,0,0,0}, y1 = {0,0,0,0,0,0,0,0}, y2 = {0,0,0,0,0,0,0,0};

#pragma unroll
        for (int ks = 0; ks < 8; ++ks) {
            const int k0 = ks * 32;
            FragB ah0, al0, ah1, al1, bh, bl;
            const unsigned short* p;
            p = v_hi + arow0 + k0 + kbaseA;  ah0.u[0] = *(const v4u*)(p);  ah0.u[1] = *(const v4u*)(p + 16);
            p = v_lo + arow0 + k0 + kbaseA;  al0.u[0] = *(const v4u*)(p);  al0.u[1] = *(const v4u*)(p + 16);
            p = v_hi + arow1 + k0 + kbaseA;  ah1.u[0] = *(const v4u*)(p);  ah1.u[1] = *(const v4u*)(p + 16);
            p = v_lo + arow1 + k0 + kbaseA;  al1.u[0] = *(const v4u*)(p);  al1.u[1] = *(const v4u*)(p + 16);
            p = &t_hi[n][k0 + kbaseB];       bh.u[0]  = *(const v4u*)(p);  bh.u[1]  = *(const v4u*)(p + 16);
            p = &t_lo[n][k0 + kbaseB];       bl.u[0]  = *(const v4u*)(p);  bl.u[1]  = *(const v4u*)(p + 16);

            // S = hi*hi + lo*hi + hi*lo ; 6 independent accumulation chains
            x0 = __builtin_amdgcn_wmma_f32_16x16x32_bf16(false, ah0.v, false, bh.v, (short)0, x0, false, false);
            y0 = __builtin_amdgcn_wmma_f32_16x16x32_bf16(false, ah1.v, false, bh.v, (short)0, y0, false, false);
            x1 = __builtin_amdgcn_wmma_f32_16x16x32_bf16(false, al0.v, false, bh.v, (short)0, x1, false, false);
            y1 = __builtin_amdgcn_wmma_f32_16x16x32_bf16(false, al1.v, false, bh.v, (short)0, y1, false, false);
            x2 = __builtin_amdgcn_wmma_f32_16x16x32_bf16(false, ah0.v, false, bl.v, (short)0, x2, false, false);
            y2 = __builtin_amdgcn_wmma_f32_16x16x32_bf16(false, ah1.v, false, bl.v, (short)0, y2, false, false);
        }
        v8f accx = x0 + x1 + x2;
        v8f accy = y0 + y1 + y2;

        epilogue_tile(accx, rt0, labs0, half, n, ld, denom, sumpos, nposv);
        epilogue_tile(accy, rt1, labs1, half, n, ld, denom, sumpos, nposv);
    }
}

// --------------------------------------------------------------- finalize ---
__global__ __launch_bounds__(512) void finalize(const float* __restrict__ denom,
                                                const float* __restrict__ sumpos,
                                                const float* __restrict__ nposv,
                                                float* __restrict__ out) {
    __shared__ float red[512];
    int i = threadIdx.x;
    red[i] = logf(denom[i]) - sumpos[i] / nposv[i];
    __syncthreads();
    for (int s = 256; s > 0; s >>= 1) {
        if (i < s) red[i] += red[i + s];
        __syncthreads();
    }
    if (i == 0) out[0] = red[0] * (1.0f / (float)B_ROWS);
}

// ----------------------------------------------------------------- launch ---
extern "C" void kernel_launch(void* const* d_in, const int* in_sizes, int n_in,
                              void* d_out, int out_size, void* d_ws, size_t ws_size,
                              hipStream_t stream) {
    (void)in_sizes; (void)n_in; (void)out_size; (void)ws_size;
    const float* visual = (const float*)d_in[0];
    const float* text   = (const float*)d_in[1];
    const int*   labels = (const int*)d_in[2];
    const int*   ids    = (const int*)d_in[3];
    // d_in[4] = d_space (compile-time D_DIM = 256)

    char* ws = (char*)d_ws;
    unsigned short* v_hi = (unsigned short*)(ws);               // 512*256*2 = 262144 B
    unsigned short* v_lo = (unsigned short*)(ws + 262144);      // 262144 B
    unsigned char*  mask = (unsigned char*)(ws + 524288);       // 50000 B (pad to 50048)
    float* accums = (float*)(ws + 574336);                      // 3*512 floats
    float* denom  = accums;
    float* sumpos = accums + B_ROWS;
    float* nposv  = accums + 2 * B_ROWS;

    init_ws   <<<196, 256, 0, stream>>>(mask, accums);
    scatter_ids<<<(N_IDS + 255) / 256, 256, 0, stream>>>(ids, mask);
    prep_visual<<<B_ROWS / 16, 256, 0, stream>>>(visual, v_hi, v_lo);
    gemm_loss <<<CT_TILES, 256, 0, stream>>>(v_hi, v_lo, text, labels, mask,
                                             denom, sumpos, nposv);
    finalize  <<<1, 512, 0, stream>>>(denom, sumpos, nposv, (float*)d_out);
}
